// SelfAttention_54271206752575
// MI455X (gfx1250) — matrix operand order repaired and checked by reading, not compiled, not run on hardware
//
#include <hip/hip_runtime.h>

// ---------------- types ----------------
typedef __attribute__((ext_vector_type(16))) __bf16 v16bf;
typedef __attribute__((ext_vector_type(8)))  __bf16 v8bf;
typedef __attribute__((ext_vector_type(8)))  float  v8f;

static __device__ __forceinline__ unsigned short f2bf(float f) {
    union { float f; unsigned u; } c; c.f = f;
    unsigned u = c.u;
    u = u + 0x7FFFu + ((u >> 16) & 1u);   // round-to-nearest-even
    return (unsigned short)(u >> 16);
}

static __device__ __forceinline__ v16bf cat8(v8bf lo, v8bf hi) {
    return __builtin_shufflevector(lo, hi, 0,1,2,3,4,5,6,7,8,9,10,11,12,13,14,15);
}

// A-fragment load: row-major row pointer, contiguous k = (lane>>4)*16 + s (k0 = (lane>>4)*8)
static __device__ __forceinline__ v16bf ldA(const unsigned short* row, int k0) {
    v8bf lo = *(const v8bf*)(row + 2 * k0);
    v8bf hi = *(const v8bf*)(row + 2 * k0 + 8);
    return cat8(lo, hi);
}

static __device__ __forceinline__ v8f wmma_bf16(v16bf a, v16bf b, v8f c) {
    return __builtin_amdgcn_wmma_f32_16x16x32_bf16(false, a, false, b, (short)0, c, false, false);
}

// LDS byte-offset of a __shared__ object (generic -> AS3 -> int)
static __device__ __forceinline__ unsigned lds_off(const void* p) {
    return (unsigned)(size_t)(__attribute__((address_space(3))) const char*)(const char*)p;
}

// CDNA5 async global->LDS copy, 16 bytes, tracked by ASYNCcnt
static __device__ __forceinline__ void async_ld_b128(const void* gp, unsigned lofs) {
    asm volatile("global_load_async_to_lds_b128 %0, %1, off"
                 :: "v"(lofs), "v"((unsigned long long)(size_t)gp) : "memory");
}
static __device__ __forceinline__ void wait_async0() {
    asm volatile("s_wait_asynccnt 0x0" ::: "memory");
}

// ---------------- constants ----------------
#define BB   4
#define SS   2048
#define CC   1024
#define HH   16
#define HD   64
#define MM   (BB * SS)        // 8192
#define NQKV 3072

// ---------------- kernel 1: fp32 -> bf16 convert ----------------
__global__ void k_cvt(const float* __restrict__ in, unsigned short* __restrict__ out, int n) {
    int i = blockIdx.x * blockDim.x + threadIdx.x;
    if (i < n) out[i] = f2bf(in[i]);
}

// ---------------- kernel 2: LayerNorm -> bf16 ----------------
__global__ void k_ln(const float* __restrict__ x, const float* __restrict__ w,
                     const float* __restrict__ b, unsigned short* __restrict__ h) {
    int row = blockIdx.x;                       // token index 0..8191
    const float* xr = x + (size_t)row * CC;
    float s = 0.f, s2 = 0.f;
    for (int i = threadIdx.x; i < CC; i += 256) { float v = xr[i]; s += v; s2 += v * v; }
    for (int off = 16; off; off >>= 1) { s += __shfl_xor(s, off, 32); s2 += __shfl_xor(s2, off, 32); }
    __shared__ float aS[8], aS2[8];
    int wid = threadIdx.x >> 5;
    if ((threadIdx.x & 31) == 0) { aS[wid] = s; aS2[wid] = s2; }
    __syncthreads();
    float ts = 0.f, ts2 = 0.f;
    #pragma unroll
    for (int i = 0; i < 8; ++i) { ts += aS[i]; ts2 += aS2[i]; }
    float mu  = ts * (1.0f / CC);
    float var = ts2 * (1.0f / CC) - mu * mu;
    float inv = rsqrtf(var + 1e-5f);
    unsigned short* hr = h + (size_t)row * CC;
    for (int i = threadIdx.x; i < CC; i += 256)
        hr[i] = f2bf((xr[i] - mu) * inv * w[i] + b[i]);
}

// ---------------- pipelined 16x128 GEMM core (shared by k_qkv / k_oproj) ----------------
// A: [M][CC] row-major bf16, B: weights [N][CC] row-major bf16 (= B-operand ready).
// Double-buffered software pipeline: fragment loads issued one full stage ahead.
static __device__ __forceinline__ void gemm16x128(const unsigned short* __restrict__ arow,
                                                  const unsigned short* __restrict__ brow,
                                                  int k0, v8f acc[8]) {
    v16bf b0[8], b1[8], a0, a1;
    #pragma unroll
    for (int t = 0; t < 8; ++t)
        b0[t] = *(const v16bf*)(brow + (size_t)t * 16 * CC);
    a0 = ldA(arow, k0);
    int kk = 0;
    for (; kk < CC - 64; kk += 64) {
        a1 = ldA(arow + kk + 32, k0);
        __builtin_prefetch(brow + kk + 128, 0, 3);
        __builtin_prefetch(arow + kk + 128, 0, 3);
        #pragma unroll
        for (int t = 0; t < 8; ++t) {
            acc[t] = wmma_bf16(a0, b0[t], acc[t]);
            b1[t] = *(const v16bf*)(brow + (size_t)t * 16 * CC + kk + 32);
        }
        a0 = ldA(arow + kk + 64, k0);
        #pragma unroll
        for (int t = 0; t < 8; ++t) {
            acc[t] = wmma_bf16(a1, b1[t], acc[t]);
            b0[t] = *(const v16bf*)(brow + (size_t)t * 16 * CC + kk + 64);
        }
    }
    // tail: kk == CC-64, no loads past end
    a1 = ldA(arow + kk + 32, k0);
    #pragma unroll
    for (int t = 0; t < 8; ++t) {
        acc[t] = wmma_bf16(a0, b0[t], acc[t]);
        b1[t] = *(const v16bf*)(brow + (size_t)t * 16 * CC + kk + 32);
    }
    #pragma unroll
    for (int t = 0; t < 8; ++t)
        acc[t] = wmma_bf16(a1, b1[t], acc[t]);
}

// ---------------- kernel 3: QKV GEMM + scatter to Q / K / V^T ----------------
// grid (MM/16, NQKV/128), block 32. Wave computes 16x128 output strip.
__global__ void __launch_bounds__(32, 2)
k_qkv(const unsigned short* __restrict__ hb, const unsigned short* __restrict__ wq,
      unsigned short* __restrict__ Qb, unsigned short* __restrict__ Kb,
      unsigned short* __restrict__ Vt) {
    const int l  = threadIdx.x;
    const int m0 = blockIdx.x * 16;
    const int n0 = blockIdx.y * 128;
    const int r = l & 15, hs = l >> 4, k0 = hs * 8;
    v8f acc[8] = {};
    const unsigned short* arow = hb + (size_t)(m0 + r) * CC;
    const unsigned short* brow = wq + (size_t)(n0 + r) * CC + hs * 16;
    gemm16x128(arow, brow, k0, acc);

    const int bidx = m0 >> 11;          // batch
    const int sb   = m0 & (SS - 1);     // row base within batch
    #pragma unroll
    for (int t = 0; t < 8; ++t) {
        int o  = n0 + t * 16 + r;       // output channel of this lane's column
        int hh = o / 192;
        int j  = o % 192;
        int bh = bidx * HH + hh;
        if (j < 128) {                  // Q or K, row-major [bh][s][d]
            unsigned short* dst = (j < 64)
                ? (Qb + ((size_t)bh * SS + sb) * HD + j)
                : (Kb + ((size_t)bh * SS + sb) * HD + (j - 64));
            #pragma unroll
            for (int i = 0; i < 8; ++i)
                dst[(size_t)(i + 8 * hs) * HD] = f2bf(acc[t][i]);
        } else {                        // V transposed [bh][d][s] : contiguous 8 over s
            int d = j - 128;
            unsigned short* dst = Vt + ((size_t)bh * HD + d) * SS + sb + 8 * hs;
            unsigned pk0 = (unsigned)f2bf(acc[t][0]) | ((unsigned)f2bf(acc[t][1]) << 16);
            unsigned pk1 = (unsigned)f2bf(acc[t][2]) | ((unsigned)f2bf(acc[t][3]) << 16);
            unsigned pk2 = (unsigned)f2bf(acc[t][4]) | ((unsigned)f2bf(acc[t][5]) << 16);
            unsigned pk3 = (unsigned)f2bf(acc[t][6]) | ((unsigned)f2bf(acc[t][7]) << 16);
            *reinterpret_cast<uint4*>(dst) = make_uint4(pk0, pk1, pk2, pk3);
        }
    }
}

// ---------------- kernel 4: flash attention ----------------
// 4 waves/block cooperate on the SAME (b,h). K/V tiles are staged into
// double-buffered LDS with async copies (ASYNCcnt) overlapping compute.
__global__ void __launch_bounds__(128, 2)
k_attn(const unsigned short* __restrict__ Qb, const unsigned short* __restrict__ Kb,
       const unsigned short* __restrict__ Vt, unsigned short* __restrict__ ctx) {
    __shared__ unsigned short ldsK[2][32 * 64];     // [buf][key][d]  2x4 KB
    __shared__ unsigned short ldsV[2][64 * 32];     // [buf][d][key]  2x4 KB
    __shared__ unsigned short ldsP[4][16 * 32];     // per-wave P       4 KB
    const int tid = threadIdx.x;
    const int wid = tid >> 5;
    const int l   = tid & 31;
    const int bh  = blockIdx.x >> 5;                // 0..63
    const int qt  = (blockIdx.x & 31) * 4 + wid;    // 0..127
    const int r = l & 15, hs = l >> 4, k0 = hs * 8;

    const unsigned short* qrow = Qb + ((size_t)bh * SS + qt * 16 + r) * HD;
    v16bf aQ0 = ldA(qrow, k0);          // d 0..31 chunk
    v16bf aQ1 = ldA(qrow + 32, k0);     // d 32..63 chunk

    v8f o0 = {}, o1 = {}, o2 = {}, o3 = {};
    float mrow[8], lrow[8];
    #pragma unroll
    for (int i = 0; i < 8; ++i) { mrow[i] = -1e30f; lrow[i] = 0.f; }
    const float scale = 0.03125f;       // 1/sqrt(1024)
    const unsigned short* kbase = Kb + (size_t)bh * SS * HD;
    const unsigned short* vbase = Vt + (size_t)bh * HD * SS;
    unsigned short* P = ldsP[wid];
    const int vd = tid >> 1, vh = (tid & 1) * 16;   // V staging coords

    // stage helper (as lambda): async copy K tile (contiguous 4KB) + V^T tile
    auto stage = [&](int kb, int buf) {
        const unsigned short* gK = kbase + (size_t)kb * HD + tid * 16;
        unsigned ko = lds_off(&ldsK[buf][0]);
        async_ld_b128(gK,     ko + tid * 32);
        async_ld_b128(gK + 8, ko + tid * 32 + 16);
        const unsigned short* gV = vbase + (size_t)vd * SS + kb + vh;
        unsigned vo = lds_off(&ldsV[buf][0]);
        async_ld_b128(gV,     vo + (vd * 32 + vh) * 2);
        async_ld_b128(gV + 8, vo + (vd * 32 + vh) * 2 + 16);
    };

    stage(0, 0);
    wait_async0();
    __syncthreads();

    for (int kb = 0; kb < SS; kb += 32) {
        const int cur = (kb >> 5) & 1;
        if (kb + 32 < SS) {
            stage(kb + 32, cur ^ 1);                    // overlaps with compute below
            __builtin_prefetch(kbase + (size_t)(kb + 64) * HD + tid * 16, 0, 3);
        }
        const unsigned short* cK = ldsK[cur];
        const unsigned short* cV = ldsV[cur];

        // ---- scores: two 16x16 tiles (keys kb..+15, kb+16..+31), K-dim = 64 ----
        v16bf kf00 = *(const v16bf*)(cK + r * 64 + hs * 16);
        v16bf kf01 = *(const v16bf*)(cK + r * 64 + 32 + hs * 16);
        v16bf kf10 = *(const v16bf*)(cK + (16 + r) * 64 + hs * 16);
        v16bf kf11 = *(const v16bf*)(cK + (16 + r) * 64 + 32 + hs * 16);
        v8f s0 = {}, s1 = {};
        s0 = wmma_bf16(aQ0, kf00, s0);
        s0 = wmma_bf16(aQ1, kf01, s0);
        s1 = wmma_bf16(aQ0, kf10, s1);
        s1 = wmma_bf16(aQ1, kf11, s1);

        // ---- online softmax (row stats across 16 lanes of each half) ----
        float corr[8];
        #pragma unroll
        for (int i = 0; i < 8; ++i) {
            float t0 = s0[i] * scale, t1 = s1[i] * scale;
            float cm = fmaxf(t0, t1);
            #pragma unroll
            for (int off = 8; off; off >>= 1) cm = fmaxf(cm, __shfl_xor(cm, off, 16));
            float mn = fmaxf(mrow[i], cm);
            float c  = __expf(mrow[i] - mn);
            float p0 = __expf(t0 - mn), p1 = __expf(t1 - mn);
            float ps = p0 + p1;
            #pragma unroll
            for (int off = 8; off; off >>= 1) ps += __shfl_xor(ps, off, 16);
            lrow[i] = lrow[i] * c + ps;
            mrow[i] = mn;
            corr[i] = c;
            s0[i] = p0; s1[i] = p1;
        }
        #pragma unroll
        for (int i = 0; i < 8; ++i) {
            o0[i] *= corr[i]; o1[i] *= corr[i]; o2[i] *= corr[i]; o3[i] *= corr[i];
        }

        // ---- relayout P (C/D f32 layout -> bf16 A layout) via wave-local LDS ----
        #pragma unroll
        for (int i = 0; i < 8; ++i) {
            int m = i + 8 * hs;
            P[m * 32 + r]      = f2bf(s0[i]);
            P[m * 32 + 16 + r] = f2bf(s1[i]);
        }
        __builtin_amdgcn_wave_barrier();     // DS ops are in-order within a wave
        v16bf pf = ldA(P + r * 32, k0);

        // ---- O += P(16x32) x V(32x64): V^T rows give contiguous B-fragments ----
        v16bf vf0 = *(const v16bf*)(cV + (0 * 16 + r) * 32 + hs * 16);
        v16bf vf1 = *(const v16bf*)(cV + (1 * 16 + r) * 32 + hs * 16);
        v16bf vf2 = *(const v16bf*)(cV + (2 * 16 + r) * 32 + hs * 16);
        v16bf vf3 = *(const v16bf*)(cV + (3 * 16 + r) * 32 + hs * 16);
        o0 = wmma_bf16(pf, vf0, o0);
        o1 = wmma_bf16(pf, vf1, o1);
        o2 = wmma_bf16(pf, vf2, o2);
        o3 = wmma_bf16(pf, vf3, o3);

        wait_async0();       // our stage writes to the other buffer are done
        __syncthreads();     // all waves done reading cur / staging complete
    }

    // ---- normalize and store context [b,s,c] bf16 ----
    const int b = bh >> 4, h = bh & 15;
    float inv[8];
    #pragma unroll
    for (int i = 0; i < 8; ++i) inv[i] = 1.0f / lrow[i];
    #pragma unroll
    for (int i = 0; i < 8; ++i) {
        int m = i + 8 * hs;
        size_t base = (size_t)(b * SS + qt * 16 + m) * CC + h * HD + r;
        ctx[base + 0]  = f2bf(o0[i] * inv[i]);
        ctx[base + 16] = f2bf(o1[i] * inv[i]);
        ctx[base + 32] = f2bf(o2[i] * inv[i]);
        ctx[base + 48] = f2bf(o3[i] * inv[i]);
    }
}

// ---------------- kernel 5: out-projection GEMM + bias + residual ----------------
// grid (MM/16, CC/128), block 32.
__global__ void __launch_bounds__(32, 2)
k_oproj(const unsigned short* __restrict__ ctx, const unsigned short* __restrict__ wo,
        const float* __restrict__ bout, const float* __restrict__ x,
        float* __restrict__ out) {
    const int l  = threadIdx.x;
    const int m0 = blockIdx.x * 16;
    const int n0 = blockIdx.y * 128;
    const int r = l & 15, hs = l >> 4, k0 = hs * 8;
    v8f acc[8] = {};
    const unsigned short* arow = ctx + (size_t)(m0 + r) * CC;
    const unsigned short* brow = wo + (size_t)(n0 + r) * CC + hs * 16;
    gemm16x128(arow, brow, k0, acc);

    #pragma unroll
    for (int t = 0; t < 8; ++t) {
        int o = n0 + t * 16 + r;
        float bo = bout[o];
        #pragma unroll
        for (int i = 0; i < 8; ++i) {
            size_t row = (size_t)(m0 + i + 8 * hs);
            out[row * CC + o] = acc[t][i] + bo + x[row * CC + o];
        }
    }
}

// ---------------- launch ----------------
extern "C" void kernel_launch(void* const* d_in, const int* in_sizes, int n_in,
                              void* d_out, int out_size, void* d_ws, size_t ws_size,
                              hipStream_t stream) {
    const float* x    = (const float*)d_in[0];
    const float* ln_w = (const float*)d_in[1];
    const float* ln_b = (const float*)d_in[2];
    const float* wqkv = (const float*)d_in[3];
    const float* wout = (const float*)d_in[4];
    const float* bout = (const float*)d_in[5];
    float* out = (float*)d_out;

    char* ws = (char*)d_ws;
    unsigned short* h_bf  = (unsigned short*)(ws);                       // 16 MB
    unsigned short* wq_bf = (unsigned short*)(ws + 16777216);            //  6 MB
    unsigned short* wo_bf = (unsigned short*)(ws + 23068672);            //  2 MB
    unsigned short* Qb    = (unsigned short*)(ws + 25165824);            // 16 MB
    unsigned short* Kb    = (unsigned short*)(ws + 41943040);            // 16 MB
    unsigned short* Vt    = (unsigned short*)(ws + 58720256);            // 16 MB
    unsigned short* ctx   = (unsigned short*)(ws + 75497472);            // 16 MB

    k_cvt<<<(NQKV * CC + 255) / 256, 256, 0, stream>>>(wqkv, wq_bf, NQKV * CC);
    k_cvt<<<(CC * CC + 255) / 256, 256, 0, stream>>>(wout, wo_bf, CC * CC);
    k_ln<<<MM, 256, 0, stream>>>(x, ln_w, ln_b, h_bf);
    k_qkv<<<dim3(MM / 16, NQKV / 128), 32, 0, stream>>>(h_bf, wq_bf, Qb, Kb, Vt);
    k_attn<<<(BB * HH * (SS / 16)) / 4, 128, 0, stream>>>(Qb, Kb, Vt, ctx);
    k_oproj<<<dim3(MM / 16, CC / 128), 32, 0, stream>>>(ctx, wo_bf, bout, x, out);
}
